// DMPNNEncoder_57243324121245
// MI455X (gfx1250) — matrix operand order repaired and checked by reading, not compiled, not run on hardware
//
#include <hip/hip_runtime.h>
#include <stdint.h>

#define N_NODES 50000
#define N_UND   400000
#define E_DIR   (2 * N_UND)
#define HIDDEN  128
#define NODE_F  133
#define EDGE_F  14
#define NUM_GRAPHS 512

#define K1   (NODE_F + EDGE_F)   // 147
#define K1P  160                 // pad K to multiple of 32
#define K2P  128
#define K3   (NODE_F + HIDDEN)   // 261
#define K3P  288

// LDS row strides (elements): +8 bf16 = +16B to rotate banks (keeps 16B align)
#define A1LD (K1P + 8)           // 168
#define A2LD (K2P + 8)           // 136
#define A3LD (K3P + 8)           // 296

#define TPB  8                   // M-tiles (of 64 rows) per block for gemm1/2

typedef __bf16    v16bf __attribute__((ext_vector_type(16)));
typedef float     v8f   __attribute__((ext_vector_type(8)));
typedef uint32_t  v4u   __attribute__((ext_vector_type(4)));
typedef unsigned short u16;

union FragAB { v16bf v; v4u q[2]; };

static __device__ __forceinline__ u16 f2bf(float f) {
    union { float f; uint32_t u; } c; c.f = f;
    uint32_t r = c.u + 0x7FFFu + ((c.u >> 16) & 1u);   // round-to-nearest-even
    return (u16)(r >> 16);
}
static __device__ __forceinline__ float bf2f(u16 h) {
    union { uint32_t u; float f; } c; c.u = ((uint32_t)h) << 16;
    return c.f;
}

// Low 32 bits of a flat pointer into the LDS aperture == LDS byte offset.
static __device__ __forceinline__ unsigned lds_off32(const void* p) {
    return (unsigned)(uintptr_t)p;
}
// CDNA5 async global->LDS copy, 16B per lane, tracked by ASYNCcnt.
static __device__ __forceinline__ void async_ld_lds_b128(unsigned lds_off, const void* gaddr) {
    asm volatile("global_load_async_to_lds_b128 %0, %1, off"
                 :: "v"(lds_off), "v"(gaddr) : "memory");
}
static __device__ __forceinline__ void wait_async0() {
    asm volatile("s_wait_asynccnt 0" ::: "memory");
}

// ---------------------------------------------------------------------------
// Transpose + pad an f32 weight [K][128] into bf16 W^T [128][Kpad]
// ---------------------------------------------------------------------------
__global__ void wt_transpose_pad(const float* __restrict__ W, u16* __restrict__ Wt,
                                 int K, int Kpad) {
    int idx = blockIdx.x * blockDim.x + threadIdx.x;
    int total = HIDDEN * Kpad;
    if (idx >= total) return;
    int n = idx / Kpad, k = idx - n * Kpad;
    float v = (k < K) ? W[k * HIDDEN + n] : 0.0f;
    Wt[idx] = f2bf(v);
}

// ---------------------------------------------------------------------------
// Load one N-tile's B fragments for all K-chunks into registers (from global
// W^T [128][KPAD]).  B per lane: col = lane&15; elems 0..15 = K[kc*32+16*half..].
// ---------------------------------------------------------------------------
template <int KPAD>
static __device__ __forceinline__ void load_B_regs(
        const u16* __restrict__ Wt, int ntile, int lane, FragAB b[KPAD / 32]) {
    const int half = lane >> 4, l15 = lane & 15;
    #pragma unroll
    for (int kc = 0; kc < KPAD / 32; ++kc) {
        const u16* brow = Wt + (ntile * 16 + l15) * KPAD + kc * 32 + 16 * half;
        b[kc].q[0] = *(const v4u*)(brow);
        b[kc].q[1] = *(const v4u*)(brow + 8);
    }
}

// A fragment from LDS: row = lane&15 of the mtile; elems 0..15 = K[kc*32+16*half..]
// (contiguous K order, matching B).
template <int LDA>
static __device__ __forceinline__ FragAB load_A_frag(
        const u16* __restrict__ A, int mtile, int kc, int lane) {
    const int half = lane >> 4, l15 = lane & 15;
    const u16* arow = A + (mtile * 16 + l15) * LDA + kc * 32;
    FragAB fa;
    fa.q[0] = *(const v4u*)(arow + 16 * half);
    fa.q[1] = *(const v4u*)(arow + 16 * half + 8);
    return fa;
}

static __device__ __forceinline__ v8f wmma_bf16(FragAB a, FragAB b, v8f c) {
    return __builtin_amdgcn_wmma_f32_16x16x32_bf16(
        false, a.v, false, b.v, (short)0, c, false, false);
}

// ---------------------------------------------------------------------------
// GEMM1: h0[e,:] = relu([x[src[e]] || edge_attr[e]] @ W1)   (bf16 out)
// Wave owns 2 N-tiles (B resident in regs for whole block), loops 4 M-tiles.
// ---------------------------------------------------------------------------
__global__ void __launch_bounds__(128)
gemm1_edge_init(const float* __restrict__ x, const float* __restrict__ ea,
                const u16* __restrict__ W1t, const int* __restrict__ src,
                u16* __restrict__ h0) {
    __shared__ __align__(16) u16 Alds[64 * A1LD];
    __shared__ int srcIds[64];
    const int tid = threadIdx.x;
    const int w = tid >> 5, lane = tid & 31;
    const int half = lane >> 4, l15 = lane & 15;

    FragAB b0[K1P / 32], b1[K1P / 32];
    load_B_regs<K1P>(W1t, 2 * w + 0, lane, b0);
    load_B_regs<K1P>(W1t, 2 * w + 1, lane, b1);

    for (int it = 0; it < TPB; ++it) {
        const long e0 = ((long)blockIdx.x * TPB + it) * 64;
        if (e0 >= E_DIR) break;
        if (tid < 64) { long e = e0 + tid; srcIds[tid] = (e < E_DIR) ? src[e] : 0; }
        __syncthreads();
        for (int idx = tid; idx < 64 * K1P; idx += 128) {
            int r = idx / K1P, k = idx - r * K1P;
            long e = e0 + r;
            float v = 0.0f;
            if (e < E_DIR) {
                if (k < NODE_F)  v = x[(long)srcIds[r] * NODE_F + k];
                else if (k < K1) v = ea[e * EDGE_F + (k - NODE_F)];
            }
            Alds[r * A1LD + k] = f2bf(v);
        }
        __syncthreads();

        v8f acc[4][2];
        #pragma unroll
        for (int m = 0; m < 4; ++m)
            #pragma unroll
            for (int n = 0; n < 2; ++n)
                #pragma unroll
                for (int j = 0; j < 8; ++j) acc[m][n][j] = 0.0f;

        #pragma unroll
        for (int m = 0; m < 4; ++m)
            #pragma unroll
            for (int kc = 0; kc < K1P / 32; ++kc) {
                FragAB fa = load_A_frag<A1LD>(Alds, m, kc, lane);
                acc[m][0] = wmma_bf16(fa, b0[kc], acc[m][0]);
                acc[m][1] = wmma_bf16(fa, b1[kc], acc[m][1]);
            }

        #pragma unroll
        for (int m = 0; m < 4; ++m)
            #pragma unroll
            for (int n = 0; n < 2; ++n)
                #pragma unroll
                for (int j = 0; j < 8; ++j) {
                    long e = e0 + m * 16 + j + 8 * half;
                    if (e < E_DIR) {
                        float v = acc[m][n][j];
                        h0[e * HIDDEN + w * 32 + n * 16 + l15] =
                            f2bf(v > 0.0f ? v : 0.0f);
                    }
                }
        __syncthreads();   // Alds reuse next iteration
    }
}

// ---------------------------------------------------------------------------
// segment_sum: node_in[dst[e],k] += h[e,k]  — 16B packed load, 8 f32 atomics
// ---------------------------------------------------------------------------
__global__ void __launch_bounds__(256)
scatter_add(const u16* __restrict__ h, const int* __restrict__ dst,
            float* __restrict__ node_in) {
    long t = (long)blockIdx.x * 256 + threadIdx.x;
    if (t >= (long)E_DIR * (HIDDEN / 8)) return;
    int e  = (int)(t >> 4);
    int k0 = ((int)t & 15) * 8;
    v4u pack = *(const v4u*)(h + (long)e * HIDDEN + k0);
    const u16* hp = (const u16*)&pack;
    float* drow = node_in + (long)dst[e] * HIDDEN + k0;
    #pragma unroll
    for (int i = 0; i < 8; ++i) atomicAdd(drow + i, bf2f(hp[i]));
}

// ---------------------------------------------------------------------------
// GEMM2: hOut[e,:] = relu(h0[e,:] + (node_in[src[e]] - hPrev[rev[e]]) @ W2)
// B resident in regs; hPrev[rev] rows gathered into LDS by the async engine.
// ---------------------------------------------------------------------------
__global__ void __launch_bounds__(128)
gemm2_update(const float* __restrict__ node_in, const u16* __restrict__ hPrev,
             const u16* __restrict__ h0, const u16* __restrict__ W2t,
             const int* __restrict__ src, const int* __restrict__ rev,
             u16* __restrict__ hOut) {
    __shared__ __align__(16) u16 Alds[64 * A2LD];
    __shared__ __align__(16) u16 Hrev[64 * A2LD];
    __shared__ int srcIds[64];
    __shared__ int revIds[64];
    const int tid = threadIdx.x;
    const int w = tid >> 5, lane = tid & 31;
    const int half = lane >> 4, l15 = lane & 15;

    FragAB b0[K2P / 32], b1[K2P / 32];
    load_B_regs<K2P>(W2t, 2 * w + 0, lane, b0);
    load_B_regs<K2P>(W2t, 2 * w + 1, lane, b1);

    const unsigned hrevBase = lds_off32(Hrev);
    const char* hPrevBytes = (const char*)hPrev;

    for (int it = 0; it < TPB; ++it) {
        const long e0 = ((long)blockIdx.x * TPB + it) * 64;
        if (e0 >= E_DIR) break;
        if (tid < 64) {
            long e = e0 + tid;
            srcIds[tid] = (e < E_DIR) ? src[e] : 0;
            revIds[tid] = (e < E_DIR) ? rev[e] : 0;
        }
        __syncthreads();

        // async-gather the 64 reverse-edge bf16 rows into LDS (16B chunks)
        for (int c = tid; c < 64 * 16; c += 128) {
            int r = c >> 4, q = c & 15;
            async_ld_lds_b128(hrevBase + (unsigned)(r * (A2LD * 2) + q * 16),
                              hPrevBytes + (long)revIds[r] * (HIDDEN * 2) + q * 16);
        }
        wait_async0();
        __syncthreads();

        // A = bf16(node_in[src] - Hrev)
        for (int idx = tid; idx < 64 * K2P; idx += 128) {
            int r = idx >> 7, k = idx & 127;
            long e = e0 + r;
            float v = 0.0f;
            if (e < E_DIR)
                v = node_in[(long)srcIds[r] * HIDDEN + k] - bf2f(Hrev[r * A2LD + k]);
            Alds[r * A2LD + k] = f2bf(v);
        }
        __syncthreads();

        v8f acc[4][2];
        #pragma unroll
        for (int m = 0; m < 4; ++m)
            #pragma unroll
            for (int n = 0; n < 2; ++n)
                #pragma unroll
                for (int j = 0; j < 8; ++j) acc[m][n][j] = 0.0f;

        #pragma unroll
        for (int m = 0; m < 4; ++m)
            #pragma unroll
            for (int kc = 0; kc < K2P / 32; ++kc) {
                FragAB fa = load_A_frag<A2LD>(Alds, m, kc, lane);
                acc[m][0] = wmma_bf16(fa, b0[kc], acc[m][0]);
                acc[m][1] = wmma_bf16(fa, b1[kc], acc[m][1]);
            }

        #pragma unroll
        for (int m = 0; m < 4; ++m)
            #pragma unroll
            for (int n = 0; n < 2; ++n)
                #pragma unroll
                for (int j = 0; j < 8; ++j) {
                    long e = e0 + m * 16 + j + 8 * half;
                    if (e < E_DIR) {
                        long o = e * HIDDEN + w * 32 + n * 16 + l15;
                        float v = bf2f(h0[o]) + acc[m][n][j];
                        hOut[o] = f2bf(v > 0.0f ? v : 0.0f);
                    }
                }
        __syncthreads();   // Alds/Hrev reuse next iteration
    }
}

// ---------------------------------------------------------------------------
// GEMM3: out[batch[n],:] += relu([x[n] || vmsg[n]] @ W3)
// (runs once over 782 blocks; wave per M-tile, B fragments from global)
// ---------------------------------------------------------------------------
__global__ void __launch_bounds__(128)
gemm3_readout(const float* __restrict__ x, const float* __restrict__ vmsg,
              const u16* __restrict__ W3t, const int* __restrict__ batch,
              float* __restrict__ out) {
    __shared__ __align__(16) u16 Alds[64 * A3LD];
    __shared__ int batchIds[64];
    const int  tid = threadIdx.x;
    const long n0  = (long)blockIdx.x * 64;
    const int w = tid >> 5, lane = tid & 31;
    const int half = lane >> 4, l15 = lane & 15;

    if (tid < 64) { long n = n0 + tid; batchIds[tid] = (n < N_NODES) ? batch[n] : 0; }
    __syncthreads();

    for (int idx = tid; idx < 64 * K3P; idx += 128) {
        int r = idx / K3P, k = idx - r * K3P;
        long n = n0 + r;
        float v = 0.0f;
        if (n < N_NODES) {
            if (k < NODE_F)  v = x[n * NODE_F + k];
            else if (k < K3) v = vmsg[n * HIDDEN + (k - NODE_F)];
        }
        Alds[r * A3LD + k] = f2bf(v);
    }
    __syncthreads();

    v8f acc[8];
    #pragma unroll
    for (int i = 0; i < 8; ++i)
        #pragma unroll
        for (int j = 0; j < 8; ++j) acc[i][j] = 0.0f;

    #pragma unroll
    for (int kc = 0; kc < K3P / 32; ++kc) {
        FragAB fa = load_A_frag<A3LD>(Alds, w, kc, lane);
        #pragma unroll
        for (int nt = 0; nt < 8; ++nt) {
            const u16* brow = W3t + (nt * 16 + l15) * K3P + kc * 32 + 16 * half;
            FragAB fb;
            fb.q[0] = *(const v4u*)(brow);
            fb.q[1] = *(const v4u*)(brow + 8);
            acc[nt] = wmma_bf16(fa, fb, acc[nt]);
        }
    }

    #pragma unroll
    for (int nt = 0; nt < 8; ++nt)
        #pragma unroll
        for (int j = 0; j < 8; ++j) {
            int  m = w * 16 + j + 8 * half;
            long n = n0 + m;
            if (n < N_NODES) {
                float v = acc[nt][j];
                v = v > 0.0f ? v : 0.0f;
                atomicAdd(&out[(long)batchIds[m] * HIDDEN + nt * 16 + l15], v);
            }
        }
}

// ---------------------------------------------------------------------------
// Host-side orchestration
// ---------------------------------------------------------------------------
static inline size_t alignup(size_t v) { return (v + 255) & ~(size_t)255; }

extern "C" void kernel_launch(void* const* d_in, const int* in_sizes, int n_in,
                              void* d_out, int out_size, void* d_ws, size_t ws_size,
                              hipStream_t stream) {
    const float* x    = (const float*)d_in[0];
    const float* ea   = (const float*)d_in[1];
    const float* W1   = (const float*)d_in[2];
    const float* W2   = (const float*)d_in[3];
    const float* W3   = (const float*)d_in[4];
    const int*   eidx = (const int*)d_in[5];
    const int*   rev  = (const int*)d_in[6];
    const int*   bat  = (const int*)d_in[7];
    const int* src = eidx;
    const int* dst = eidx + E_DIR;
    float* out = (float*)d_out;

    // workspace carve-up
    char* p = (char*)d_ws;
    u16* W1t = (u16*)p;  p += alignup((size_t)HIDDEN * K1P * 2);
    u16* W2t = (u16*)p;  p += alignup((size_t)HIDDEN * K2P * 2);
    u16* W3t = (u16*)p;  p += alignup((size_t)HIDDEN * K3P * 2);
    u16* h0  = (u16*)p;  p += alignup((size_t)E_DIR * HIDDEN * 2);
    u16* hA  = (u16*)p;  p += alignup((size_t)E_DIR * HIDDEN * 2);
    u16* hB  = (u16*)p;  p += alignup((size_t)E_DIR * HIDDEN * 2);
    float* node_in = (float*)p;  // N_NODES*HIDDEN f32
    (void)ws_size; (void)in_sizes; (void)n_in; (void)out_size;

    const size_t nodeInBytes = (size_t)N_NODES * HIDDEN * sizeof(float);

    hipMemsetAsync(d_out, 0, (size_t)NUM_GRAPHS * HIDDEN * sizeof(float), stream);

    wt_transpose_pad<<<(HIDDEN * K1P + 255) / 256, 256, 0, stream>>>(W1, W1t, K1, K1P);
    wt_transpose_pad<<<(HIDDEN * K2P + 255) / 256, 256, 0, stream>>>(W2, W2t, HIDDEN, K2P);
    wt_transpose_pad<<<(HIDDEN * K3P + 255) / 256, 256, 0, stream>>>(W3, W3t, K3, K3P);

    const int nTiles     = (E_DIR + 63) / 64;               // 12500
    const int edgeBlocks = (nTiles + TPB - 1) / TPB;        // 1563
    const int scatBlocks = (int)(((long)E_DIR * (HIDDEN / 8) + 255) / 256);
    const int nodeBlocks = (N_NODES + 63) / 64;             // 782

    // h0 = relu([x[src] || ea] @ W1)
    gemm1_edge_init<<<edgeBlocks, 128, 0, stream>>>(x, ea, W1t, src, h0);

    // iteration 1: h = relu(h0 + (seg(h0)[src] - h0[rev]) @ W2)
    hipMemsetAsync(node_in, 0, nodeInBytes, stream);
    scatter_add<<<scatBlocks, 256, 0, stream>>>(h0, dst, node_in);
    gemm2_update<<<edgeBlocks, 128, 0, stream>>>(node_in, h0, h0, W2t, src, rev, hA);

    // iteration 2
    hipMemsetAsync(node_in, 0, nodeInBytes, stream);
    scatter_add<<<scatBlocks, 256, 0, stream>>>(hA, dst, node_in);
    gemm2_update<<<edgeBlocks, 128, 0, stream>>>(node_in, hA, h0, W2t, src, rev, hB);

    // v_msg = segment_sum(h_final)
    hipMemsetAsync(node_in, 0, nodeInBytes, stream);
    scatter_add<<<scatBlocks, 256, 0, stream>>>(hB, dst, node_in);

    // out[batch] += relu([x || v_msg] @ W3)
    gemm3_readout<<<nodeBlocks, 128, 0, stream>>>(x, node_in, W3t, bat, out);
}
